// GE2ELoss_44908178047448
// MI455X (gfx1250) — compile-verified, matrix operand order checked by reading, not run on hardware
//
#include <hip/hip_runtime.h>
#include <hip/hip_bf16.h>
#include <math.h>

typedef __attribute__((ext_vector_type(16))) _Float16 v16h;
typedef __attribute__((ext_vector_type(8)))  float    v8f;

#define D_DIM 256          // embedding dim (fixed by reference)
#define S_DIM 1024         // num speakers (fixed by reference)
#define NCHUNK 128         // centroid columns staged in LDS per chunk
#define LDS_STRIDE 272     // halves per column: 256 + 16 pad -> 544B, 32B aligned
#define ROWS_PER_WG 128    // 8 waves x 16 rows
#define NCHUNKS (S_DIM / NCHUNK)

// ---------------- kernel 0: zero accumulators + output ----------------
__global__ void k_init(float* __restrict__ segsum, float* __restrict__ counts,
                       float* __restrict__ out) {
    int i = blockIdx.x * blockDim.x + threadIdx.x;
    if (i < S_DIM * D_DIM) segsum[i] = 0.0f;
    if (i < S_DIM)         counts[i] = 0.0f;
    if (i == 0)            out[0]    = 0.0f;
}

// ---------------- kernel 1: segment-sum via fp32 global atomics ----------------
__global__ void k_accum(const float* __restrict__ emb, const int* __restrict__ sid,
                        float* __restrict__ segsum, float* __restrict__ counts, int B) {
    int idx = blockIdx.x * blockDim.x + threadIdx.x;  // over B*D
    int b = idx >> 8;
    int d = idx & (D_DIM - 1);
    if (b < B) {
        int s = sid[b];
        atomicAdd(&segsum[s * D_DIM + d], emb[idx]);
        if (d == 0) atomicAdd(&counts[s], 1.0f);
    }
}

// ---------------- kernel 2: centroid mean + L2 normalize -> f16 ----------------
__global__ void k_centroid(const float* __restrict__ segsum, const float* __restrict__ counts,
                           _Float16* __restrict__ centh) {
    __shared__ float red[D_DIM];
    int s = blockIdx.x;
    int d = threadIdx.x;
    float cnt = fmaxf(counts[s], 1.0f);
    float v = segsum[s * D_DIM + d] / cnt;
    red[d] = v * v;
    __syncthreads();
    for (int off = D_DIM / 2; off > 0; off >>= 1) {
        if (d < off) red[d] += red[d + off];
        __syncthreads();
    }
    float inv = 1.0f / fmaxf(sqrtf(red[0]), 1e-12f);
    centh[s * D_DIM + d] = (_Float16)(v * inv);
}

// ---- kernel 3: fused WMMA GEMM + bounded-shift softmax + CE reduce ----
__global__ __launch_bounds__(256, 1) void k_loss(
    const float* __restrict__ emb, const _Float16* __restrict__ centh,
    const int* __restrict__ sid, const float* __restrict__ wp,
    const float* __restrict__ bp, float* __restrict__ out, int B) {

    __shared__ _Float16 lds[NCHUNK * LDS_STRIDE];  // ~68 KB of 320 KB WGP LDS

    const int tid  = threadIdx.x;
    const int wave = tid >> 5;
    const int lane = tid & 31;
    const int hlf  = lane >> 4;   // 0: lanes 0-15, 1: lanes 16-31
    const int lcol = lane & 15;
    const int rowBase = blockIdx.x * ROWS_PER_WG + wave * 16;
    const int myrow = rowBase + lcol;

    const float w  = wp[0];
    const float bb = bp[0];
    // logits are w*cos+b with cos in [-1,1] => bounded above by |w|+b
    const float shift = fabsf(w) + bb;

    // --- A fragments: 16 rows x 256 K, fp32 -> f16 in registers (64 VGPRs) ---
    v16h a[8];
    {
        const float* rp = emb + (size_t)myrow * D_DIM;
        #pragma unroll
        for (int kk = 0; kk < 8; ++kk) {
            const float* q = rp + kk * 32 + hlf * 16;
            v16h t;
            #pragma unroll
            for (int j = 0; j < 16; ++j) t[j] = (_Float16)q[j];
            a[kk] = t;
        }
    }

    // labels for the 8 C-matrix rows this lane owns (M = hlf*8 + r)
    int lab[8];
    #pragma unroll
    for (int r = 0; r < 8; ++r) lab[r] = sid[rowBase + hlf * 8 + r];

    // per-lane state: exp-sum (fixed shift) + harvested label logit
    float msum[8], mlab[8];
    #pragma unroll
    for (int r = 0; r < 8; ++r) { msum[r] = 0.0f; mlab[r] = 0.0f; }

    #pragma unroll 1
    for (int chunk = 0; chunk < NCHUNKS; ++chunk) {
        __syncthreads();  // previous chunk fully consumed
        // --- async stage of 128 centroid columns into LDS (ASYNCcnt path) ---
        // NOTE: the `lds` pointer is passed as a (template-unused) asm input so
        // the shared object ESCAPES into the asm; otherwise LLVM proves the asm
        // cannot write it and deletes the later ds_load_b128 B-fragment reads.
        const _Float16* src = centh + (size_t)chunk * NCHUNK * D_DIM;
        for (int i = tid; i < NCHUNK * 32; i += 256) {   // 16B pieces, 32 per column
            int col = i >> 5, seg = i & 31;
            unsigned loff = (unsigned)(col * (LDS_STRIDE * 2) + seg * 16); // LDS byte offset
            const _Float16* g = src + col * D_DIM + seg * 8;
            asm volatile("global_load_async_to_lds_b128 %0, %1, off"
                         :: "v"(loff), "v"(g), "v"((const void*)lds) : "memory");
        }
        if (chunk + 1 < NCHUNKS)  // hint next chunk toward L2 (global_prefetch_b8)
            __builtin_prefetch(centh + (size_t)(chunk + 1) * NCHUNK * D_DIM + tid * 16, 0, 1);
        asm volatile("s_wait_asynccnt 0x0" :: "v"((const void*)lds) : "memory"); // drain my async loads
        __syncthreads();                                  // whole chunk visible

        #pragma unroll 1
        for (int t = 0; t < NCHUNK / 16; ++t) {
            v8f c = {};
            const _Float16* bcol = lds + (t * 16 + lcol) * LDS_STRIDE + hlf * 16;
            #pragma unroll
            for (int kk = 0; kk < 8; ++kk) {
                v16h bf = *(const v16h*)(bcol + kk * 32);
                c = __builtin_amdgcn_wmma_f32_16x16x32_f16(
                        /*neg_a=*/false, a[kk], /*neg_b=*/false, bf,
                        /*c_mod=*/(short)0, c, /*reuse_a=*/false, /*reuse_b=*/false);
            }
            const int gcol = chunk * NCHUNK + t * 16 + lcol;
            #pragma unroll
            for (int r = 0; r < 8; ++r) {
                float v = fmaf(w, c[r], bb);          // logit (row hlf*8+r, col gcol)
                msum[r] += __expf(v - shift);         // bounded: v - shift <= 0
                if (lab[r] == gcol) mlab[r] = v;      // harvest label logit
            }
        }
    }

    // --- plain-sum merge across the 16 lanes of each half-wave ---
    #pragma unroll
    for (int r = 0; r < 8; ++r) {
        #pragma unroll
        for (int off = 1; off < 16; off <<= 1) {
            msum[r] += __shfl_xor(msum[r], off, 32);
            mlab[r] += __shfl_xor(mlab[r], off, 32);  // exactly one lane contributed
        }
    }

    if (lcol == 0) {
        float acc = 0.0f;
        #pragma unroll
        for (int r = 0; r < 8; ++r)
            acc += (shift + __logf(msum[r])) - mlab[r];   // -logp[label]
        atomicAdd(out, acc * (1.0f / (float)B));
    }
}

// ---------------- host-side launcher ----------------
extern "C" void kernel_launch(void* const* d_in, const int* in_sizes, int n_in,
                              void* d_out, int out_size, void* d_ws, size_t ws_size,
                              hipStream_t stream) {
    const float* emb = (const float*)d_in[0];
    const float* w   = (const float*)d_in[1];
    const float* b   = (const float*)d_in[2];
    const int*   sid = (const int*)d_in[3];
    // d_in[4] = num_speakers (device int scalar) — shapes are fixed, S_DIM=1024

    const int B = in_sizes[0] / D_DIM;   // 131072

    float*    segsum = (float*)d_ws;                      // S*D fp32   (1 MB)
    float*    counts = segsum + S_DIM * D_DIM;            // S fp32
    _Float16* centh  = (_Float16*)(counts + S_DIM);       // S*D f16    (512 KB)
    float*    out    = (float*)d_out;

    k_init<<<(S_DIM * D_DIM + 255) / 256, 256, 0, stream>>>(segsum, counts, out);
    k_accum<<<(unsigned)((size_t)B * D_DIM / 256), 256, 0, stream>>>(emb, sid, segsum, counts, B);
    k_centroid<<<S_DIM, D_DIM, 0, stream>>>(segsum, counts, centh);
    k_loss<<<B / ROWS_PER_WG, 256, 0, stream>>>(emb, centh, sid, w, b, out, B);
}
